// HDSuperpositionEmbedding_32762010534134
// MI455X (gfx1250) — matrix operand order repaired and checked by reading, not compiled, not run on hardware
//
#include <hip/hip_runtime.h>

// ---------------------------------------------------------------------------
// Types for CDNA5 WMMA (wave32): bf16 A/B fragments (16 halves = 8 VGPRs),
// f32 C/D accumulator (8 VGPRs).
// ---------------------------------------------------------------------------
typedef __attribute__((ext_vector_type(16))) __bf16 bfx16;
typedef __attribute__((ext_vector_type(8)))  float  fx8;

#define DIM_D   1024
#define DIM_DQ  256
#define NBR     4
#define VOCAB   32000
#define NTOK    16384   // B*S = 8*2048

__device__ __forceinline__ fx8 wmma_bf16(bfx16 a, bfx16 b, fx8 c) {
    // D = A(16x32 bf16) * B(32x16 bf16) + C(16x16 f32)
    return __builtin_amdgcn_wmma_f32_16x16x32_bf16(
        false, a, false, b, (short)0, c, false, false);
}

// ---------------------------------------------------------------------------
// Fragment loaders
// ---------------------------------------------------------------------------
// Raw f32 A-row fetch (issue loads early for software pipelining).
// A fragment (16x32, 16-bit): lane<16 holds M=lane, K = k0+{0..7, 16..23};
// lane>=16 holds M=lane-16, K = k0+{8..15, 24..31}.
struct RawA { float4 x0, x1, x2, x3; };

__device__ __forceinline__ RawA fetch_row_f32(const float* __restrict__ row,
                                              int k0, int lane) {
    int ks = k0 + ((lane & 16) ? 8 : 0);
    const float4* p0 = reinterpret_cast<const float4*>(row + ks);
    const float4* p1 = reinterpret_cast<const float4*>(row + ks + 16);
    RawA r;
    r.x0 = p0[0]; r.x1 = p0[1]; r.x2 = p1[0]; r.x3 = p1[1];
    return r;
}

__device__ __forceinline__ bfx16 cvt_scale(const RawA& f, float s) {
    bfx16 r;
    r[0]=(__bf16)(f.x0.x*s);  r[1]=(__bf16)(f.x0.y*s);  r[2]=(__bf16)(f.x0.z*s);  r[3]=(__bf16)(f.x0.w*s);
    r[4]=(__bf16)(f.x1.x*s);  r[5]=(__bf16)(f.x1.y*s);  r[6]=(__bf16)(f.x1.z*s);  r[7]=(__bf16)(f.x1.w*s);
    r[8]=(__bf16)(f.x2.x*s);  r[9]=(__bf16)(f.x2.y*s);  r[10]=(__bf16)(f.x2.z*s); r[11]=(__bf16)(f.x2.w*s);
    r[12]=(__bf16)(f.x3.x*s); r[13]=(__bf16)(f.x3.y*s); r[14]=(__bf16)(f.x3.z*s); r[15]=(__bf16)(f.x3.w*s);
    return r;
}

// A fragment from a bf16 row (same layout): two 16B loads.
__device__ __forceinline__ bfx16 load_a_bf16(const __bf16* __restrict__ row,
                                             int k0, int lane) {
    int ks = k0 + ((lane & 16) ? 8 : 0);
    union { uint4 q[2]; bfx16 v; } u;
    u.q[0] = *reinterpret_cast<const uint4*>(row + ks);
    u.q[1] = *reinterpret_cast<const uint4*>(row + ks + 16);
    return u.v;
}

// B fragment (32x16, 16-bit): lane<16 holds N=lane, K=k0+{0..7, 16..23};
// lane>=16 holds N=lane-16, K=k0+{8..15, 24..31}.  WT is N-major bf16 [N][K].
__device__ __forceinline__ bfx16 load_b_bf16(const __bf16* __restrict__ WT,
                                             int colBase, int k0, int lane, int K) {
    int n  = colBase + (lane & 15);
    int kb = k0 + ((lane & 16) ? 8 : 0);
    const uint4* p = reinterpret_cast<const uint4*>(WT + (size_t)n * K + kb);
    union { uint4 q[2]; bfx16 v; } u;
    u.q[0] = p[0]; u.q[1] = p[2];
    return u.v;
}

// B fragment from an LDS slab with 40-element (80 B) row pitch.
__device__ __forceinline__ bfx16 load_b_lds(const __bf16 (*buf)[40], int colLocal,
                                            int lane) {
    int n  = colLocal + (lane & 15);
    int kb = (lane & 16) ? 8 : 0;
    const uint4* p = reinterpret_cast<const uint4*>(&buf[n][kb]);
    union { uint4 q[2]; bfx16 v; } u;
    u.q[0] = p[0]; u.q[1] = p[2];
    return u.v;
}

// ---------------------------------------------------------------------------
// gfx1250 async copy: 16B global -> LDS, tracked by ASYNCcnt.
// ---------------------------------------------------------------------------
__device__ __forceinline__ void async_cp16(const __bf16* __restrict__ g,
                                           const __bf16* l) {
    unsigned loff = (unsigned)(size_t)(const void*)l;  // LDS aperture low bits
    asm volatile("global_load_async_to_lds_b128 %0, %1, off"
                 :: "v"(loff), "v"(g) : "memory");
}
#define WAIT_ASYNC0() asm volatile("s_wait_asynccnt 0x0" ::: "memory")

// ---------------------------------------------------------------------------
// Kernel 0: per-branch modulation factor = 0.9 + 0.2*sigmoid(mean(basis[n]))
// ---------------------------------------------------------------------------
__global__ void factors_kernel(const float* __restrict__ bb, float* __restrict__ fac) {
    __shared__ float red[NBR][64];
    int n = threadIdx.x >> 6, l = threadIdx.x & 63;
    float s = 0.f;
    for (int i = l; i < DIM_D; i += 64) s += bb[n * DIM_D + i];
    red[n][l] = s;
    __syncthreads();
    if (l == 0) {
        float tot = 0.f;
        for (int i = 0; i < 64; ++i) tot += red[n][i];
        float m = tot * (1.0f / DIM_D);
        fac[n] = 0.9f + 0.2f / (1.0f + __expf(-m));
    }
}

// ---------------------------------------------------------------------------
// Kernel 1: transpose + f32->bf16 convert:  dst[c][r] = src[r][c]
// ---------------------------------------------------------------------------
__global__ void transpose_cvt(const float* __restrict__ src, __bf16* __restrict__ dst,
                              int R, int C) {
    int idx = blockIdx.x * blockDim.x + threadIdx.x;
    int total = R * C;
    for (; idx < total; idx += gridDim.x * blockDim.x) {
        int r = idx / C, c = idx - r * C;
        dst[(size_t)c * R + r] = (__bf16)src[idx];
    }
}

// ---------------------------------------------------------------------------
// Kernel 2: fused gather -> q/k projection (WMMA bf16) -> scores -> softmax
//           -> collapse.  16 tokens per block, 256 threads (8 waves).
// The HBM-critical embedding gathers are software-pipelined one K-chunk ahead.
// ---------------------------------------------------------------------------
__global__ __launch_bounds__(256) void qk_attn_collapse(
    const int*   __restrict__ token_ids,   // [NTOK][4]
    const float* __restrict__ emb,         // [VOCAB][1024]
    const float* __restrict__ bq,          // [256]
    const float* __restrict__ bk,          // [256]
    const __bf16* __restrict__ WqT,        // [256][1024] bf16
    const __bf16* __restrict__ WkT,        // [256][1024] bf16
    const float* __restrict__ factors,     // [4]
    __bf16*      __restrict__ collapsed)   // [NTOK][1024] bf16
{
    __shared__ __bf16 kS[64][DIM_DQ + 8];   // ~33.8 KB
    __shared__ float  qS[16][DIM_DQ + 4];   // ~16.6 KB
    __shared__ float  scoreS[64];
    __shared__ float  attnS[16][NBR];
    __shared__ float  facS[NBR];

    const int tid  = threadIdx.x;
    const int w    = tid >> 5;
    const int lane = tid & 31;
    const int T0   = blockIdx.x * 16;

    if (tid < NBR) facS[tid] = factors[tid];
    __syncthreads();

    // --- per-lane gather rows ---------------------------------------------
    const int rg = w & 3;                     // k row-group of this wave
    const int Rk = rg * 16 + (lane & 15);     // k row 0..63
    const int tK = Rk >> 2, nK = Rk & 3;
    int idK = token_ids[(size_t)(T0 + tK) * NBR + nK];
    idK = idK < 0 ? 0 : (idK > VOCAB - 1 ? VOCAB - 1 : idK);
    const float* rowK = emb + (size_t)idK * DIM_D;
    const float  fK   = facS[nK];

    int idQ = token_ids[(size_t)(T0 + (lane & 15)) * NBR + 0];
    idQ = idQ < 0 ? 0 : (idQ > VOCAB - 1 ? VOCAB - 1 : idQ);
    const float* rowQ = emb + (size_t)idQ * DIM_D;

    const int cgK = (w >> 2) * 8;   // k col-group base (8 tiles)
    const int cgQ = 2 * w;          // q col-group base (2 tiles)

    fx8 accK[8], accQ[2];
#pragma unroll
    for (int j = 0; j < 8; ++j) accK[j] = (fx8)0.f;
#pragma unroll
    for (int j = 0; j < 2; ++j) accQ[j] = (fx8)0.f;

    // --- K-loop: pipelined gathers, bf16 WMMA, f32 accumulate -------------
    RawA rK = fetch_row_f32(rowK, 0, lane);
    RawA rQ = fetch_row_f32(rowQ, 0, lane);
    for (int k0 = 0; k0 < DIM_D; k0 += 32) {
        RawA rKn, rQn;
        if (k0 + 32 < DIM_D) {                     // prefetch next chunk
            rKn = fetch_row_f32(rowK, k0 + 32, lane);
            rQn = fetch_row_f32(rowQ, k0 + 32, lane);
        }
        bfx16 aK = cvt_scale(rK, fK);     // mod = emb * factor
        bfx16 aQ = cvt_scale(rQ, 1.0f);   // query = raw branch-0 emb
#pragma unroll
        for (int j = 0; j < 8; ++j) {
            bfx16 b = load_b_bf16(WkT, (cgK + j) * 16, k0, lane, DIM_D);
            accK[j] = wmma_bf16(aK, b, accK[j]);
        }
#pragma unroll
        for (int j = 0; j < 2; ++j) {
            bfx16 b = load_b_bf16(WqT, (cgQ + j) * 16, k0, lane, DIM_D);
            accQ[j] = wmma_bf16(aQ, b, accQ[j]);
        }
        if (k0 + 32 < DIM_D) { rK = rKn; rQ = rQn; }
    }

    // --- store q/k tiles (+bias) to LDS.  C layout: VGPR v -> row v (+8 if
    //     lane>=16), col = lane&15. -----------------------------------------
    const int rlo  = (lane & 16) ? 8 : 0;
    const int col0 = lane & 15;
#pragma unroll
    for (int j = 0; j < 8; ++j) {
        int cb = (cgK + j) * 16 + col0;
        float bkv = bk[cb];
#pragma unroll
        for (int v = 0; v < 8; ++v)
            kS[rg * 16 + rlo + v][cb] = (__bf16)(accK[j][v] + bkv);
    }
#pragma unroll
    for (int j = 0; j < 2; ++j) {
        int cb = (cgQ + j) * 16 + col0;
        float bqv = bq[cb];
#pragma unroll
        for (int v = 0; v < 8; ++v)
            qS[rlo + v][cb] = accQ[j][v] + bqv;
    }
    __syncthreads();

    // --- scores = (k . q) / sqrt(256) -------------------------------------
    if (tid < 64) {
        int t = tid >> 2;
        float s = 0.f;
        for (int c = 0; c < DIM_DQ; ++c) s += (float)kS[tid][c] * qS[t][c];
        scoreS[tid] = s * 0.0625f;
    }
    __syncthreads();

    // --- softmax over 4 branches; fold branch factor into weight ----------
    if (tid < 16) {
        float s0 = scoreS[tid * 4 + 0], s1 = scoreS[tid * 4 + 1];
        float s2 = scoreS[tid * 4 + 2], s3 = scoreS[tid * 4 + 3];
        float mx = fmaxf(fmaxf(s0, s1), fmaxf(s2, s3));
        float e0 = __expf(s0 - mx), e1 = __expf(s1 - mx);
        float e2 = __expf(s2 - mx), e3 = __expf(s3 - mx);
        float inv = 1.0f / (e0 + e1 + e2 + e3);
        attnS[tid][0] = e0 * inv * facS[0];
        attnS[tid][1] = e1 * inv * facS[1];
        attnS[tid][2] = e2 * inv * facS[2];
        attnS[tid][3] = e3 * inv * facS[3];
    }
    __syncthreads();

    // --- collapse: sum_n attn*factor * emb_row_n ; write bf16 -------------
    {
        int t = tid >> 4;
        int dbase = (tid & 15) * 64;
        const float* r[NBR];
        float wgt[NBR];
#pragma unroll
        for (int n = 0; n < NBR; ++n) {
            int id = token_ids[(size_t)(T0 + t) * NBR + n];
            id = id < 0 ? 0 : (id > VOCAB - 1 ? VOCAB - 1 : id);
            r[n] = emb + (size_t)id * DIM_D;
            wgt[n] = attnS[t][n];
        }
        __bf16* outRow = collapsed + (size_t)(T0 + t) * DIM_D;
        for (int d = dbase; d < dbase + 64; d += 4) {
            float ax = 0.f, ay = 0.f, az = 0.f, aw = 0.f;
#pragma unroll
            for (int n = 0; n < NBR; ++n) {
                float4 x = *reinterpret_cast<const float4*>(r[n] + d);
                ax += wgt[n] * x.x; ay += wgt[n] * x.y;
                az += wgt[n] * x.z; aw += wgt[n] * x.w;
            }
            union { __bf16 h[4]; uint2 u; } pk;
            pk.h[0] = (__bf16)ax; pk.h[1] = (__bf16)ay;
            pk.h[2] = (__bf16)az; pk.h[3] = (__bf16)aw;
            *reinterpret_cast<uint2*>(outRow + d) = pk.u;
        }
    }
}

// ---------------------------------------------------------------------------
// Kernel 3: out = collapsed(bf16) @ Wo + bo    (16384x1024 @ 1024x1024, f32 out)
// Block tile 64 x 128, 256 threads / 8 waves.
// B slab (128 cols x 32 k = 8 KB) staged in LDS via async-to-LDS, ping-pong
// double buffered; A fragments register double-buffered.  One
// s_wait_asynccnt 0 + barrier per chunk; WMMA stream never waits on global.
// ---------------------------------------------------------------------------
__global__ __launch_bounds__(256) void out_gemm(
    const __bf16* __restrict__ A,     // [NTOK][1024] bf16
    const __bf16* __restrict__ WoT,   // [1024 outcol][1024 k] bf16
    const float*  __restrict__ bo,    // [1024]
    float*        __restrict__ out)   // [NTOK][1024]
{
    __shared__ __bf16 bufB[2][128][40];   // 80 B pitch (16B aligned, bank-spread)

    const int tid  = threadIdx.x;
    const int w    = tid >> 5;
    const int lane = tid & 31;
    const int Mbase = blockIdx.x * 64;
    const int Nbase = blockIdx.y * 128;
    const int rg  = w & 3;
    const int cg0 = (w >> 2) * 4;

    const __bf16* arow = A + (size_t)(Mbase + rg * 16 + (lane & 15)) * DIM_D;

    fx8 acc[4];
#pragma unroll
    for (int j = 0; j < 4; ++j) acc[j] = (fx8)0.f;

    // issue one 8 KB chunk: 512 x 16B transfers, 2 per thread
    auto issue_B = [&](int k0, __bf16 (*buf)[40]) {
#pragma unroll
        for (int j = 0; j < 2; ++j) {
            int lin = tid * 2 + j;        // 0..511
            int col = lin >> 2;           // 0..127
            int q   = lin & 3;            // 16B quadrant of the 64B row segment
            async_cp16(WoT + (size_t)(Nbase + col) * DIM_D + k0 + q * 8,
                       &buf[col][q * 8]);
        }
    };

    constexpr int NC = DIM_D / 32;

    issue_B(0, bufB[0]);
    bfx16 a_cur = load_a_bf16(arow, 0, lane);
    WAIT_ASYNC0();
    __syncthreads();                       // buf0 ready for all waves

    for (int c = 0; c < NC; ++c) {
        const int p = c & 1;
        bfx16 a_next;
        if (c + 1 < NC) {                  // overlap next slab with compute
            issue_B((c + 1) * 32, bufB[p ^ 1]);
            a_next = load_a_bf16(arow, (c + 1) * 32, lane);
        }
#pragma unroll
        for (int j = 0; j < 4; ++j) {
            bfx16 b = load_b_lds(bufB[p], (cg0 + j) * 16, lane);
            acc[j] = wmma_bf16(a_cur, b, acc[j]);
        }
        if (c + 1 < NC) {
            a_cur = a_next;
            WAIT_ASYNC0();                 // my async loads for buf p^1 done
            __syncthreads();               // everyone's done -> safe to read
        }
    }

    const int rlo  = (lane & 16) ? 8 : 0;
    const int col0 = lane & 15;
#pragma unroll
    for (int j = 0; j < 4; ++j) {
        int col = Nbase + (cg0 + j) * 16 + col0;
        float bov = bo[col];
#pragma unroll
        for (int v = 0; v < 8; ++v)
            out[(size_t)(Mbase + rg * 16 + rlo + v) * DIM_D + col] = acc[j][v] + bov;
    }
}

// ---------------------------------------------------------------------------
// Host-side launcher
// ---------------------------------------------------------------------------
extern "C" void kernel_launch(void* const* d_in, const int* in_sizes, int n_in,
                              void* d_out, int out_size, void* d_ws, size_t ws_size,
                              hipStream_t stream) {
    const int*   token_ids = (const int*)  d_in[0];  // (8,2048,4)
    const float* emb_table = (const float*)d_in[1];  // (32000,1024)
    const float* basis     = (const float*)d_in[2];  // (4,1024)
    const float* Wq        = (const float*)d_in[3];  // (1024,256)
    const float* bq        = (const float*)d_in[4];  // (256)
    const float* Wk        = (const float*)d_in[5];  // (1024,256)
    const float* bk        = (const float*)d_in[6];  // (256)
    const float* Wo        = (const float*)d_in[7];  // (1024,1024)
    const float* bo        = (const float*)d_in[8];  // (1024)
    float* out = (float*)d_out;

    // workspace layout (all 256B aligned)
    char* ws = (char*)d_ws;
    float*  d_fac  = (float*) (ws);                              // 16 B (pad 256)
    __bf16* d_WqT  = (__bf16*)(ws + 256);                        // 512 KB
    __bf16* d_WkT  = (__bf16*)(ws + 256 + 524288);               // 512 KB
    __bf16* d_WoT  = (__bf16*)(ws + 256 + 2 * 524288);           // 2 MB
    __bf16* d_coll = (__bf16*)(ws + 256 + 2 * 524288 + 2097152); // 32 MB

    factors_kernel<<<1, 256, 0, stream>>>(basis, d_fac);

    transpose_cvt<<<1024, 256, 0, stream>>>(Wq, d_WqT, DIM_D, DIM_DQ);
    transpose_cvt<<<1024, 256, 0, stream>>>(Wk, d_WkT, DIM_D, DIM_DQ);
    transpose_cvt<<<4096, 256, 0, stream>>>(Wo, d_WoT, DIM_D, DIM_D);

    qk_attn_collapse<<<NTOK / 16, 256, 0, stream>>>(
        token_ids, emb_table, bq, bk, d_WqT, d_WkT, d_fac, d_coll);

    out_gemm<<<dim3(NTOK / 64, DIM_D / 128, 1), 256, 0, stream>>>(
        d_coll, d_WoT, bo, out);
}
